// NodeEdgeNet_12017318494546
// MI455X (gfx1250) — matrix operand order repaired and checked by reading, not compiled, not run on hardware
//
#include <hip/hip_runtime.h>
#include <hip/hip_bf16.h>

typedef __attribute__((ext_vector_type(16))) _Float16 v16h;
typedef __attribute__((ext_vector_type(8)))  _Float16 v8h;
typedef __attribute__((ext_vector_type(8)))  float    v8f;

#define NNODE 50000
#define NEDGE 800000
#define ND 128
#define ED 64

struct MlpP { const _Float16* w1; const float* b1; const float* lng; const float* lnb;
              const _Float16* w2; const float* b2; };
struct FfnP { const _Float16* bond; const _Float16* node; MlpP inter; MlpP gate; };

// Per-workgroup (one wave) LDS scratch: ~32KB
struct alignas(16) WS {
  _Float16 X [16*224];  // GEMM A-stage (max K-pad 224)
  _Float16 Xb[16*128];  // secondary f16 stage
  _Float16 Xn[16*64];   // xf stage (kernel B)
  _Float16 He[16*64];   // h_edge tile f16
  float    Y [16*128];  // GEMM f32 out
  float    Y2[16*128];  // second f32 stage
  int li[16]; int ri[16];
  float t0[16]; float t1[16];
};

__device__ inline float sigm(float x) { return 1.0f / (1.0f + __expf(-x)); }

__device__ inline void atomAdd(float* p, float v) {
  __hip_atomic_fetch_add(p, v, __ATOMIC_RELAXED, __HIP_MEMORY_SCOPE_AGENT);
}

// Y[16][dout] = X(16 x kpad, f16 LDS) * Wt(f16 [dout][kpad] transposed) + bias
// WMMA f32_16x16x32_f16; A layout: lane<16: K 0..7 / 16..23, lane>=16: K 8..15 / 24..31
// B layout: lane<16: K 0..7 / 16..23, lane>=16: K 8..15 / 24..31, column = lane&15
__device__ inline void wave_gemm(const _Float16* X, int ldx,
                                 const _Float16* Wt, int kpad,
                                 const float* bias, float* Y, int ldy, int dout)
{
  const int lane = threadIdx.x & 31;
  const int half = lane >> 4;
  const int l15  = lane & 15;
  for (int n0 = 0; n0 < dout; n0 += 16) {
    v8f acc = {};
    for (int k0 = 0; k0 < kpad; k0 += 32) {
      const _Float16* ap = X + l15 * ldx + k0 + 8 * half;
      v8h a0 = *(const v8h*)ap;
      v8h a1 = *(const v8h*)(ap + 16);
      const _Float16* bp = Wt + (size_t)(n0 + l15) * kpad + k0 + 8 * half;
      v8h b0 = *(const v8h*)bp;
      v8h b1 = *(const v8h*)(bp + 16);
      v16h a, b;
#pragma unroll
      for (int i = 0; i < 8; ++i) { a[i] = a0[i]; a[i+8] = a1[i]; b[i] = b0[i]; b[i+8] = b1[i]; }
      acc = __builtin_amdgcn_wmma_f32_16x16x32_f16(false, a, false, b, (short)0, acc, false, false);
    }
    float bv = bias ? bias[n0 + l15] : 0.0f;
#pragma unroll
    for (int r = 0; r < 8; ++r)
      Y[(r + 8 * half) * ldy + n0 + l15] = acc[r] + bv;  // D: row=r+8*half, col=n0+(lane&15)
  }
}

// LayerNorm(d) + ReLU, f32 Y (packed width d) -> f16 X (packed width kpad, zero padded)
__device__ inline void ln_relu(const float* Y, int d, const float* g, const float* b,
                               _Float16* X, int kpad)
{
  int lane = threadIdx.x & 31;
  if (lane < 16) {
    const float* y = Y + lane * d;
    _Float16* x = X + lane * kpad;
    float mu = 0.f;
    for (int i = 0; i < d; ++i) mu += y[i];
    mu /= (float)d;
    float var = 0.f;
    for (int i = 0; i < d; ++i) { float t = y[i] - mu; var += t * t; }
    var /= (float)d;
    float inv = rsqrtf(var + 1e-5f);
    for (int i = 0; i < d; ++i) {
      float v = (y[i] - mu) * inv * g[i] + b[i];
      x[i] = (_Float16)(v > 0.f ? v : 0.f);
    }
    for (int i = d; i < kpad; ++i) x[i] = (_Float16)0.f;
  }
}

// prefetch a gathered row of `nf` floats (nf*4 bytes) into cache (global_prefetch_b8)
// 32 lanes cover 16 rows x 2 segments
__device__ inline void prefetch_rows(const float* base, const int* idx, int nf)
{
  const int lane = threadIdx.x & 31;
  const int r = lane >> 1, q = lane & 1;
  const float* p = base + (size_t)idx[r] * nf + q * (nf >> 1);
  __builtin_prefetch(p, 0, 3);
  if (nf >= 128) __builtin_prefetch(p + 32, 0, 3);   // rows of 128 floats = 4 cachelines
}

// convert f32 W[din][dout] -> f16 Wt[dout][kpad] (zero-pad K)
__global__ void kConv(const float* __restrict__ W, _Float16* __restrict__ Wt,
                      int din, int dout, int kpad)
{
  int i = blockIdx.x * blockDim.x + threadIdx.x;
  if (i >= dout * kpad) return;
  int n = i / kpad, k = i - n * kpad;
  Wt[i] = (k < din) ? (_Float16)W[(size_t)k * dout + n] : (_Float16)0.f;
}

// bond_ffn over an edge tile; scatter result into obuf[sidx]
__device__ void ffn_edge(WS& s, const FfnP& f, const float* hnode,
                         const int* nidx, const int* sidx, float* obuf)
{
  const int lane = threadIdx.x;
  wave_gemm(s.He, 64, f.bond, 64, nullptr, s.Y, 128, 128);
  __syncthreads();
  for (int t = lane; t < 16*128; t += 32) s.Xb[t] = (_Float16)s.Y[t];
  __syncthreads();
  for (int t = lane; t < 16*128; t += 32) { int r = t >> 7, c = t & 127;
    s.X[t] = (_Float16)hnode[(size_t)nidx[r] * ND + c]; }
  __syncthreads();
  wave_gemm(s.X, 128, f.node, 128, nullptr, s.Y, 128, 128);
  __syncthreads();
  for (int t = lane; t < 16*128; t += 32) s.X[t] = (_Float16)(s.Y[t] * (float)s.Xb[t]);
  __syncthreads();
  wave_gemm(s.X, 128, f.inter.w1, 128, f.inter.b1, s.Y, 128, 128);
  __syncthreads();
  ln_relu(s.Y, 128, f.inter.lng, f.inter.lnb, s.X, 128);
  __syncthreads();
  wave_gemm(s.X, 128, f.inter.w2, 128, f.inter.b2, s.Y2, 64, 64);
  __syncthreads();
  // gate input: [h_edge(64) | h_node(128) | edge_time(1) | pad] -> kpad 224
  for (int t = lane; t < 16*224; t += 32) {
    int r = t / 224, c = t - r * 224; _Float16 v;
    if (c < 64)        v = s.He[r * 64 + c];
    else if (c < 192)  v = (_Float16)hnode[(size_t)nidx[r] * ND + (c - 64)];
    else if (c == 192) v = (_Float16)s.t1[r];
    else               v = (_Float16)0.f;
    s.X[t] = v;
  }
  __syncthreads();
  wave_gemm(s.X, 224, f.gate.w1, 224, f.gate.b1, s.Y, 32, 32);
  __syncthreads();
  ln_relu(s.Y, 32, f.gate.lng, f.gate.lnb, s.X, 32);
  __syncthreads();
  wave_gemm(s.X, 32, f.gate.w2, 32, f.gate.b2, s.Y, 64, 64);
  __syncthreads();
  for (int t = lane; t < 16*64; t += 32) { int r = t >> 6, c = t & 63;
    atomAdd(&obuf[(size_t)sidx[r] * ED + c], s.Y2[t] * sigm(s.Y[t])); }
  __syncthreads();
}

// ----- per-node pre: hn = MLP(node_net,h), centroid, node_l/node_r feats -----
__global__ __launch_bounds__(32) void kNpre(
  const float* __restrict__ hnode, MlpP nn,
  const _Float16* wCent, const float* bCent,
  const _Float16* wNL, const float* bNL,
  const _Float16* wNR, const float* bNR,
  float* hn, float* cent, float* nlf, float* nrf)
{
  __shared__ WS s;
  const int lane = threadIdx.x;
  const int n0 = blockIdx.x * 16;
  for (int t = lane; t < 16*128; t += 32) s.Xb[t] = (_Float16)hnode[(size_t)n0 * ND + t];
  __syncthreads();
  wave_gemm(s.Xb, 128, nn.w1, 128, nn.b1, s.Y, 128, 128);
  __syncthreads();
  ln_relu(s.Y, 128, nn.lng, nn.lnb, s.X, 128);
  __syncthreads();
  wave_gemm(s.X, 128, nn.w2, 128, nn.b2, s.Y, 128, 128);
  __syncthreads();
  for (int t = lane; t < 16*128; t += 32) hn[(size_t)n0 * ND + t] = s.Y[t];
  __syncthreads();
  wave_gemm(s.Xb, 128, wCent, 128, bCent, s.Y, 128, 128);
  __syncthreads();
  for (int t = lane; t < 16*128; t += 32) cent[(size_t)n0 * ND + t] = s.Y[t];
  __syncthreads();
  wave_gemm(s.Xb, 128, wNL, 128, bNL, s.Y, 64, 64);
  __syncthreads();
  for (int t = lane; t < 16*64; t += 32) nlf[(size_t)n0 * ED + t] = s.Y[t];
  __syncthreads();
  wave_gemm(s.Xb, 128, wNR, 128, bNR, s.Y, 64, 64);
  __syncthreads();
  for (int t = lane; t < 16*64; t += 32) nrf[(size_t)n0 * ED + t] = s.Y[t];
}

// ----- per-edge A: edge_emb, node-msg+gate scatter, two bond-FFN scatters -----
__global__ __launch_bounds__(32) void kA(
  const int* __restrict__ eidx, const float* __restrict__ hnode,
  const float* __restrict__ pos, float* __restrict__ hedge,
  const float* __restrict__ ntime, const float* __restrict__ etime,
  const float* __restrict__ hn, float* aggr, float* bufL, float* bufR,
  const _Float16* wEmb, const float* bEmb,
  MlpP eNet, const _Float16* wMsg, const float* bMsg, MlpP nGate,
  FfnP fL, FfnP fR)
{
  __shared__ WS s;
  const int lane = threadIdx.x;
  const int e0 = blockIdx.x * 16;
  if (lane < 16) {
    int e = e0 + lane;
    s.li[lane] = eidx[e];
    s.ri[lane] = eidx[NEDGE + e];
    s.t0[lane] = ntime[s.ri[lane]];
    s.t1[lane] = etime[e];
  }
  __syncthreads();
  // hide random-gather latency: prefetch node rows needed much later in this tile
  prefetch_rows(hnode, s.ri, ND);
  prefetch_rows(hnode, s.li, ND);
  prefetch_rows(hn,    s.ri, ND);
  // X = [h_edge(64) | gauss(16) | pad] width 96
  for (int t = lane; t < 16*64; t += 32) { int r = t >> 6, c = t & 63;
    s.X[r * 96 + c] = (_Float16)hedge[(size_t)(e0 + r) * ED + c]; }
  if (lane < 16) {
    int r = lane;
    float rx = pos[(size_t)s.li[r]*3+0] - pos[(size_t)s.ri[r]*3+0];
    float ry = pos[(size_t)s.li[r]*3+1] - pos[(size_t)s.ri[r]*3+1];
    float rz = pos[(size_t)s.li[r]*3+2] - pos[(size_t)s.ri[r]*3+2];
    float d  = sqrtf(rx*rx + ry*ry + rz*rz);
    const float step = 10.0f / 15.0f, coeff = -0.5f / (step * step);
    for (int g = 0; g < 16; ++g) { float u = d - step * (float)g;
      s.X[r * 96 + 64 + g] = (_Float16)__expf(coeff * u * u); }
    for (int c = 80; c < 96; ++c) s.X[r * 96 + c] = (_Float16)0.f;
  }
  __syncthreads();
  wave_gemm(s.X, 96, wEmb, 96, bEmb, s.Y, 64, 64);
  __syncthreads();
  for (int t = lane; t < 16*64; t += 32) { int r = t >> 6, c = t & 63;
    float v = s.Y[t]; hedge[(size_t)(e0 + r) * ED + c] = v; s.He[t] = (_Float16)v; }
  __syncthreads();
  // edge_net MLP
  wave_gemm(s.He, 64, eNet.w1, 64, eNet.b1, s.Y, 128, 128);
  __syncthreads();
  ln_relu(s.Y, 128, eNet.lng, eNet.lnb, s.X, 128);
  __syncthreads();
  wave_gemm(s.X, 128, eNet.w2, 128, eNet.b2, s.Y, 128, 128);
  __syncthreads();
  // msg = (he * hn[right]) @ Wmsg
  for (int t = lane; t < 16*128; t += 32) { int r = t >> 7, c = t & 127;
    s.X[t] = (_Float16)(s.Y[t] * hn[(size_t)s.ri[r] * ND + c]); }
  __syncthreads();
  wave_gemm(s.X, 128, wMsg, 128, bMsg, s.Y2, 128, 128);
  __syncthreads();
  // node gate: [h_edge | h_node[right] | node_time[right] | pad] -> 224
  for (int t = lane; t < 16*224; t += 32) {
    int r = t / 224, c = t - r * 224; _Float16 v;
    if (c < 64)        v = s.He[r * 64 + c];
    else if (c < 192)  v = (_Float16)hnode[(size_t)s.ri[r] * ND + (c - 64)];
    else if (c == 192) v = (_Float16)s.t0[r];
    else               v = (_Float16)0.f;
    s.X[t] = v;
  }
  __syncthreads();
  wave_gemm(s.X, 224, nGate.w1, 224, nGate.b1, s.Y, 128, 128);
  __syncthreads();
  ln_relu(s.Y, 128, nGate.lng, nGate.lnb, s.X, 128);
  __syncthreads();
  wave_gemm(s.X, 128, nGate.w2, 128, nGate.b2, s.Y, 128, 128);
  __syncthreads();
  for (int t = lane; t < 16*128; t += 32) { int r = t >> 7, c = t & 127;
    atomAdd(&aggr[(size_t)s.li[r] * ND + c], s.Y2[t] * sigm(s.Y[t])); }
  __syncthreads();
  ffn_edge(s, fL, hnode, s.li, s.ri, bufL);   // m_l scattered by right
  ffn_edge(s, fR, hnode, s.ri, s.li, bufR);   // m_r scattered by left
}

// ----- per-node post: h_node update + pos-block left/right MLPs -----
__global__ __launch_bounds__(32) void kNpost(
  float* __restrict__ hnode, const float* __restrict__ cent, const float* __restrict__ aggr,
  const float* lng, const float* lnb, const _Float16* wOut, const float* bOut,
  MlpP pl, MlpP pr, float* lf, float* rf)
{
  __shared__ WS s;
  const int lane = threadIdx.x;
  const int n0 = blockIdx.x * 16;
  for (int t = lane; t < 16*128; t += 32)
    s.Y[t] = cent[(size_t)n0 * ND + t] + aggr[(size_t)n0 * ND + t];
  __syncthreads();
  ln_relu(s.Y, 128, lng, lnb, s.X, 128);
  __syncthreads();
  wave_gemm(s.X, 128, wOut, 128, bOut, s.Y2, 128, 128);
  __syncthreads();
  for (int t = lane; t < 16*128; t += 32) {
    float v = hnode[(size_t)n0 * ND + t] + s.Y2[t];
    hnode[(size_t)n0 * ND + t] = v;
    s.Xb[t] = (_Float16)v;
  }
  __syncthreads();
  wave_gemm(s.Xb, 128, pl.w1, 128, pl.b1, s.Y, 64, 64);
  __syncthreads();
  ln_relu(s.Y, 64, pl.lng, pl.lnb, s.X, 64);
  __syncthreads();
  wave_gemm(s.X, 64, pl.w2, 64, pl.b2, s.Y, 64, 64);
  __syncthreads();
  for (int t = lane; t < 16*64; t += 32) lf[(size_t)n0 * ED + t] = s.Y[t];
  __syncthreads();
  wave_gemm(s.Xb, 128, pr.w1, 128, pr.b1, s.Y, 64, 64);
  __syncthreads();
  ln_relu(s.Y, 64, pr.lng, pr.lnb, s.X, 64);
  __syncthreads();
  wave_gemm(s.X, 64, pr.w2, 64, pr.b2, s.Y, 64, 64);
  __syncthreads();
  for (int t = lane; t < 16*64; t += 32) rf[(size_t)n0 * ED + t] = s.Y[t];
}

// ----- per-edge B: edge update + pos-force scatter -----
__global__ __launch_bounds__(32) void kB(
  const int* __restrict__ eidx, const float* __restrict__ pos, float* __restrict__ hedge,
  const float* __restrict__ etime,
  const float* bufL, const float* bufR, const float* nlf, const float* nrf,
  const float* lf, const float* rf,
  const _Float16* wSelf, const float* bSelf,
  const float* elng, const float* elnb,
  const _Float16* wEOut, const float* bEOut,
  const _Float16* wBond, const _Float16* wNode,
  const _Float16* wI1, const float* bI1, const float* ilng, const float* ilnb,
  const float* wI2, const float* bI2,
  const _Float16* wG1, const float* bG1, const float* glng, const float* glnb,
  const float* wG2, const float* bG2,
  float* pacc)
{
  __shared__ WS s;
  const int lane = threadIdx.x;
  const int e0 = blockIdx.x * 16;
  if (lane < 16) { int e = e0 + lane;
    s.li[lane] = eidx[e]; s.ri[lane] = eidx[NEDGE + e]; s.t1[lane] = etime[e]; }
  __syncthreads();
  // prefetch all gathered per-node rows used by this tile
  prefetch_rows(bufL, s.li, ED);
  prefetch_rows(bufR, s.ri, ED);
  prefetch_rows(nlf,  s.li, ED);
  prefetch_rows(nrf,  s.ri, ED);
  prefetch_rows(lf,   s.li, ED);
  prefetch_rows(rf,   s.ri, ED);
  for (int t = lane; t < 16*64; t += 32) {
    float v = hedge[(size_t)e0 * ED + t]; s.Y2[t] = v; s.He[t] = (_Float16)v; }
  __syncthreads();
  wave_gemm(s.He, 64, wSelf, 64, bSelf, s.Y, 64, 64);
  __syncthreads();
  for (int t = lane; t < 16*64; t += 32) { int r = t >> 6, c = t & 63;
    s.Y[t] += bufL[(size_t)s.li[r] * ED + c] + bufR[(size_t)s.ri[r] * ED + c]
            + nlf [(size_t)s.li[r] * ED + c] + nrf [(size_t)s.ri[r] * ED + c]; }
  __syncthreads();
  ln_relu(s.Y, 64, elng, elnb, s.X, 64);
  __syncthreads();
  wave_gemm(s.X, 64, wEOut, 64, bEOut, s.Y, 64, 64);
  __syncthreads();
  for (int t = lane; t < 16*64; t += 32) {
    float v = s.Y2[t] + s.Y[t]; hedge[(size_t)e0 * ED + t] = v; s.He[t] = (_Float16)v; }
  __syncthreads();
  // xf = lf[left] * rf[right]
  for (int t = lane; t < 16*64; t += 32) { int r = t >> 6, c = t & 63;
    s.Xn[t] = (_Float16)(lf[(size_t)s.li[r] * ED + c] * rf[(size_t)s.ri[r] * ED + c]); }
  __syncthreads();
  wave_gemm(s.He, 64, wBond, 64, nullptr, s.Y, 128, 128);
  __syncthreads();
  for (int t = lane; t < 16*128; t += 32) s.Xb[t] = (_Float16)s.Y[t];
  __syncthreads();
  wave_gemm(s.Xn, 64, wNode, 64, nullptr, s.Y, 128, 128);
  __syncthreads();
  for (int t = lane; t < 16*128; t += 32) s.X[t] = (_Float16)(s.Y[t] * (float)s.Xb[t]);
  __syncthreads();
  wave_gemm(s.X, 128, wI1, 128, bI1, s.Y, 128, 128);
  __syncthreads();
  ln_relu(s.Y, 128, ilng, ilnb, s.X, 128);
  __syncthreads();
  if (lane < 16) { float acc = bI2[0];
    for (int c = 0; c < 128; ++c) acc += (float)s.X[lane * 128 + c] * wI2[c];
    s.t0[lane] = acc; }
  __syncthreads();
  // gate: [h_edge(64) | xf(64) | edge_time | pad] -> 160
  for (int t = lane; t < 16*160; t += 32) {
    int r = t / 160, c = t - r * 160; _Float16 v;
    if (c < 64)        v = s.He[r * 64 + c];
    else if (c < 128)  v = s.Xn[r * 64 + (c - 64)];
    else if (c == 128) v = (_Float16)s.t1[r];
    else               v = (_Float16)0.f;
    s.X[t] = v;
  }
  __syncthreads();
  wave_gemm(s.X, 160, wG1, 160, bG1, s.Y, 32, 32);
  __syncthreads();
  ln_relu(s.Y, 32, glng, glnb, s.X, 32);
  __syncthreads();
  if (lane < 16) {
    float acc = bG2[0];
    for (int c = 0; c < 32; ++c) acc += (float)s.X[lane * 32 + c] * wG2[c];
    float w = s.t0[lane] * sigm(acc);
    int l = s.li[lane], r = s.ri[lane];
    float rx = pos[(size_t)l*3+0] - pos[(size_t)r*3+0];
    float ry = pos[(size_t)l*3+1] - pos[(size_t)r*3+1];
    float rz = pos[(size_t)l*3+2] - pos[(size_t)r*3+2];
    float d  = sqrtf(rx*rx + ry*ry + rz*rz);
    float scl = w / (d * (d + 1.0f));
    atomAdd(&pacc[(size_t)l*3+0], scl * rx);
    atomAdd(&pacc[(size_t)l*3+1], scl * ry);
    atomAdd(&pacc[(size_t)l*3+2], scl * rz);
  }
}

__global__ void kPos(float* pos, const float* pacc, int n)
{
  int i = blockIdx.x * blockDim.x + threadIdx.x;
  if (i < n) pos[i] += pacc[i];
}

extern "C" void kernel_launch(void* const* d_in, const int* in_sizes, int n_in,
                              void* d_out, int out_size, void* d_ws, size_t ws_size,
                              hipStream_t stream)
{
  const float* h_node_in = (const float*)d_in[0];
  const float* pos_in    = (const float*)d_in[1];
  const float* h_edge_in = (const float*)d_in[2];
  const int*   eidx      = (const int*)  d_in[3];
  const float* ntime     = (const float*)d_in[4];
  const float* etime     = (const float*)d_in[5];

  float* out = (float*)d_out;
  float* st_node = out;                          // [NNODE][128]
  float* st_pos  = out + (size_t)NNODE * ND;     // [NNODE][3]
  float* st_edge = st_pos + (size_t)NNODE * 3;   // [NEDGE][64]

  hipMemcpyAsync(st_node, h_node_in, (size_t)NNODE * ND * 4, hipMemcpyDeviceToDevice, stream);
  hipMemcpyAsync(st_pos,  pos_in,    (size_t)NNODE * 3  * 4, hipMemcpyDeviceToDevice, stream);
  hipMemcpyAsync(st_edge, h_edge_in, (size_t)NEDGE * ED * 4, hipMemcpyDeviceToDevice, stream);

  float* f = (float*)d_ws;
  float* hn   = f; f += (size_t)NNODE * ND;
  float* cent = f; f += (size_t)NNODE * ND;
  float* aggr = f; f += (size_t)NNODE * ND;
  float* nlf  = f; f += (size_t)NNODE * ED;
  float* nrf  = f; f += (size_t)NNODE * ED;
  float* lf   = f; f += (size_t)NNODE * ED;
  float* rf   = f; f += (size_t)NNODE * ED;
  float* bufL = f; f += (size_t)NNODE * ED;
  float* bufR = f; f += (size_t)NNODE * ED;
  float* pacc = f; f += (size_t)NNODE * 4;
  _Float16* wbase = (_Float16*)f;

  const int NTe = NEDGE / 16, NTn = NNODE / 16;

  for (int blk = 0; blk < 2; ++blk) {
    void* const* P = d_in + 6 + blk * 92;   // 92 param leaves per block (sorted-dict order)
    auto Lf = [&](int i) -> const float* { return (const float*)P[i]; };
    size_t woff = 0;
    auto CW = [&](int leaf, int din, int dout) -> const _Float16* {
      int kpad = (din + 31) & ~31;
      _Float16* dst = wbase + woff; woff += (size_t)dout * kpad;
      int n = dout * kpad;
      kConv<<<(n + 255) / 256, 256, 0, stream>>>(Lf(leaf), dst, din, dout, kpad);
      return dst;
    };

    // edge_block leaves 0..37, edge_emb 38..39, node_block 40..65, pos_block 66..91
    const _Float16* wEmb = CW(39, 80, 64);
    MlpP eNet  = { CW(43, 64, 128), Lf(42), Lf(47), Lf(46), CW(45, 128, 128), Lf(44) };
    const _Float16* wMsg = CW(57, 128, 128);
    MlpP nGate = { CW(49, 193, 128), Lf(48), Lf(53), Lf(52), CW(51, 128, 128), Lf(50) };
    MlpP nn    = { CW(59, 128, 128), Lf(58), Lf(63), Lf(62), CW(61, 128, 128), Lf(60) };
    const _Float16* wCent = CW(41, 128, 128);
    const _Float16* wNOut = CW(65, 128, 128);
    FfnP fL = { CW(0, 64, 128), CW(13, 128, 128),
                { CW(8, 128, 128), Lf(7), Lf(12), Lf(11), CW(10, 128, 64), Lf(9) },
                { CW(2, 193, 32),  Lf(1), Lf(6),  Lf(5),  CW(4, 32, 64),   Lf(3) } };
    FfnP fR = { CW(14, 64, 128), CW(27, 128, 128),
                { CW(22, 128, 128), Lf(21), Lf(26), Lf(25), CW(24, 128, 64), Lf(23) },
                { CW(16, 193, 32),  Lf(15), Lf(20), Lf(19), CW(18, 32, 64),  Lf(17) } };
    const _Float16* wNL   = CW(31, 128, 64);
    const _Float16* wNR   = CW(33, 128, 64);
    const _Float16* wSelf = CW(37, 64, 64);
    const _Float16* wEOut = CW(35, 64, 64);
    MlpP pl = { CW(81, 128, 64), Lf(80), Lf(85), Lf(84), CW(83, 64, 64), Lf(82) };
    MlpP pr = { CW(87, 128, 64), Lf(86), Lf(91), Lf(90), CW(89, 64, 64), Lf(88) };
    const _Float16* wPB  = CW(66, 64, 128);
    const _Float16* wPN  = CW(79, 64, 128);
    const _Float16* wPI1 = CW(74, 128, 128);
    const _Float16* wPG1 = CW(68, 129, 32);

    hipMemsetAsync(aggr, 0, (size_t)NNODE * ND * 4, stream);
    hipMemsetAsync(bufL, 0, (size_t)NNODE * ED * 4, stream);
    hipMemsetAsync(bufR, 0, (size_t)NNODE * ED * 4, stream);
    hipMemsetAsync(pacc, 0, (size_t)NNODE * 4  * 4, stream);

    kNpre<<<NTn, 32, 0, stream>>>(st_node, nn, wCent, Lf(40), wNL, Lf(30), wNR, Lf(32),
                                  hn, cent, nlf, nrf);
    kA<<<NTe, 32, 0, stream>>>(eidx, st_node, st_pos, st_edge, ntime, etime,
                               hn, aggr, bufL, bufR,
                               wEmb, Lf(38), eNet, wMsg, Lf(56), nGate, fL, fR);
    kNpost<<<NTn, 32, 0, stream>>>(st_node, cent, aggr, Lf(55), Lf(54), wNOut, Lf(64),
                                   pl, pr, lf, rf);
    kB<<<NTe, 32, 0, stream>>>(eidx, st_pos, st_edge, etime,
                               bufL, bufR, nlf, nrf, lf, rf,
                               wSelf, Lf(36), Lf(29), Lf(28), wEOut, Lf(34),
                               wPB, wPN, wPI1, Lf(73), Lf(78), Lf(77), Lf(76), Lf(75),
                               wPG1, Lf(67), Lf(72), Lf(71), Lf(70), Lf(69),
                               pacc);
    kPos<<<(NNODE * 3 + 255) / 256, 256, 0, stream>>>(st_pos, pacc, NNODE * 3);
  }
}
